// RoPEDecoderBlock_73289321939532
// MI455X (gfx1250) — matrix operand order repaired and checked by reading, not compiled, not run on hardware
//
#include <hip/hip_runtime.h>
#include <cstdint>
#include <cstddef>

// ---------------------------------------------------------------------------
// Types for CDNA5 WMMA (wave32): bf16 16x16x32, f32 accumulate.
// ---------------------------------------------------------------------------
typedef __attribute__((ext_vector_type(16))) __bf16 v16bf;
typedef __attribute__((ext_vector_type(8)))  __bf16 v8bf;
typedef __attribute__((ext_vector_type(8)))  float  v8f;

#define B_   2
#define T_   2048
#define D_   1024
#define H_   16
#define HD_  64
#define HID_ 4096
#define ROWS (B_ * T_)

static __device__ __forceinline__ v8f wmma_bf16(v16bf a, v16bf b, v8f c) {
  // (neg_a, A, neg_b, B, c_mod, C, reuse_a, reuse_b)
  return __builtin_amdgcn_wmma_f32_16x16x32_bf16(false, a, false, b, (short)0, c,
                                                 false, false);
}

// Async global->LDS copy of 32 bytes (two b128 beats), ASYNCcnt-tracked.
static __device__ __forceinline__ void async_copy32(uint32_t lds_off,
                                                    const __bf16* gptr) {
  uint64_t ga = (uint64_t)(uintptr_t)gptr;
  asm volatile("global_load_async_to_lds_b128 %0, %1, off"
               :: "v"(lds_off), "v"(ga) : "memory");
  asm volatile("global_load_async_to_lds_b128 %0, %1, off offset:16"
               :: "v"(lds_off), "v"(ga) : "memory");
}
static __device__ __forceinline__ void wait_async0() {
  asm volatile("s_wait_asynccnt 0" ::: "memory");
}

// ---------------------------------------------------------------------------
// Weight convert + transpose (LDS-tiled, coalesced both ways):
// W f32 [K,N] row-major -> Wt bf16 [N,K] row-major.
// Grid: (N/32, K/32), block (32,8).
// ---------------------------------------------------------------------------
__global__ __launch_bounds__(256) void convT_kernel(const float* __restrict__ W,
                                                    __bf16* __restrict__ Wt,
                                                    int K, int N) {
  __shared__ __bf16 tile[32][33];
  const int n0 = blockIdx.x * 32;
  const int k0 = blockIdx.y * 32;
  const int c = threadIdx.x;
  const int r0 = threadIdx.y;
#pragma unroll
  for (int i = 0; i < 4; ++i) {
    const int r = r0 + i * 8;  // k index
    tile[r][c] = (__bf16)W[(size_t)(k0 + r) * N + n0 + c];
  }
  __syncthreads();
#pragma unroll
  for (int i = 0; i < 4; ++i) {
    const int r = r0 + i * 8;  // n index
    Wt[(size_t)(n0 + r) * K + k0 + c] = tile[c][r];
  }
}

// ---------------------------------------------------------------------------
// LayerNorm over last dim (D=1024), f32 in -> bf16 out. One block per row.
// ---------------------------------------------------------------------------
__global__ __launch_bounds__(256) void ln_kernel(const float* __restrict__ x,
                                                 const float* __restrict__ scale,
                                                 const float* __restrict__ bias,
                                                 __bf16* __restrict__ out,
                                                 int D) {
  __shared__ float red_s[8];
  __shared__ float red_ss[8];
  const int row  = blockIdx.x;
  const int tid  = threadIdx.x;
  const int lane = tid & 31;
  const int wave = tid >> 5;
  const float* xr = x + (size_t)row * D;

  float s = 0.f, ss = 0.f;
  for (int i = tid; i < D; i += 256) {
    float v = xr[i];
    s += v;
    ss += v * v;
  }
#pragma unroll
  for (int m = 1; m < 32; m <<= 1) {
    s  += __shfl_xor(s, m, 32);
    ss += __shfl_xor(ss, m, 32);
  }
  if (lane == 0) { red_s[wave] = s; red_ss[wave] = ss; }
  __syncthreads();
  float ts = 0.f, tss = 0.f;
#pragma unroll
  for (int w = 0; w < 8; ++w) { ts += red_s[w]; tss += red_ss[w]; }
  const float mu  = ts / (float)D;
  const float var = tss / (float)D - mu * mu;
  const float rs  = rsqrtf(var + 1e-6f);
  for (int i = tid; i < D; i += 256) {
    float v = (xr[i] - mu) * rs * scale[i] + bias[i];
    out[(size_t)row * D + i] = (__bf16)v;
  }
}

// ---------------------------------------------------------------------------
// RoPE: qkv bf16 [ROWS, 3*D] -> Q,K bf16 [BH, T, 64] (rotary applied).
// i is the fastest index so qkv reads and Q/K writes stay coalesced.
// ---------------------------------------------------------------------------
__global__ __launch_bounds__(256) void rope_kernel(const __bf16* __restrict__ qkv,
                                                   __bf16* __restrict__ Q,
                                                   __bf16* __restrict__ K,
                                                   int T) {
  int idx = blockIdx.x * 256 + threadIdx.x;
  const int i = idx & 31;  idx >>= 5;
  const int h = idx & 15;  idx >>= 4;
  const int t = idx % T;
  const int b = idx / T;
  if (b >= B_) return;

  const size_t row  = (size_t)b * T + t;
  const size_t base = row * (size_t)(3 * D_) + (size_t)h * HD_;

  const float invf = __expf(-logf(10000.f) * (float)(2 * i) / (float)HD_);
  const float ang  = (float)t * invf;
  float c, s;
  __sincosf(ang, &s, &c);

  const float q1 = (float)qkv[base + i];
  const float q2 = (float)qkv[base + 32 + i];
  const float k1 = (float)qkv[base + D_ + i];
  const float k2 = (float)qkv[base + D_ + 32 + i];

  const int bh = b * H_ + h;
  const size_t qk = ((size_t)bh * T + t) * HD_;
  Q[qk + i]      = (__bf16)(q1 * c - q2 * s);
  Q[qk + 32 + i] = (__bf16)(q1 * s + q2 * c);
  K[qk + i]      = (__bf16)(k1 * c - k2 * s);
  K[qk + 32 + i] = (__bf16)(k1 * s + k2 * c);
}

// ---------------------------------------------------------------------------
// V transpose (LDS-tiled): qkv[..., 2D + h*64 + d] -> Vt bf16 [BH, 64, T].
// Grid: (T/32, 2, BH), block (32,8).
// ---------------------------------------------------------------------------
__global__ __launch_bounds__(256) void vtrans_kernel(const __bf16* __restrict__ qkv,
                                                     __bf16* __restrict__ Vt,
                                                     int T) {
  __shared__ __bf16 tile[32][33];
  const int bh = blockIdx.z;
  const int b = bh >> 4, h = bh & 15;
  const int t0 = blockIdx.x * 32;
  const int d0 = blockIdx.y * 32;
  const int c = threadIdx.x;
  const int r0 = threadIdx.y;
#pragma unroll
  for (int i = 0; i < 4; ++i) {
    const int r = r0 + i * 8;  // t index
    tile[r][c] =
        qkv[(size_t)((size_t)b * T + t0 + r) * (3 * D_) + 2 * D_ + h * HD_ + d0 + c];
  }
  __syncthreads();
#pragma unroll
  for (int i = 0; i < 4; ++i) {
    const int r = r0 + i * 8;  // d index
    Vt[((size_t)bh * HD_ + d0 + r) * T + t0 + c] = tile[c][r];
  }
}

// ---------------------------------------------------------------------------
// Tiled bf16 WMMA GEMM:  C[M,N] = A[M,K] @ W[K,N], W given transposed [N,K].
// Block tile 128x128, 8 waves arranged 4(M) x 2(N): each wave computes a
// 32x64 sub-tile = 2 A-frags x 4 B-frags -> 8 WMMAs per K-step, with each
// B-fragment load feeding two back-to-back WMMAs (halves LDS-wait density
// vs a 16x128 wave tile) and only 12 ds_load_b128 per K-step.
// Double-buffered LDS staged with global_load_async_to_lds_b128 (ASYNCcnt):
// tile i+1 streams in while tile i's WMMAs execute.
// Epilogue modes: 0: bf16 | 1: f32 acc+res | 2: bf16 gelu(acc+bias)
//                 3: f32 acc+bias+res
// ---------------------------------------------------------------------------
__global__ __launch_bounds__(256) void gemm_bf16_kernel(
    const __bf16* __restrict__ A, const __bf16* __restrict__ Wt,
    int M, int N, int K, int mode,
    const float* __restrict__ bias, const float* __restrict__ res,
    float* __restrict__ outF, __bf16* __restrict__ outB) {
  __shared__ __align__(32) __bf16 As[2][128 * 32];
  __shared__ __align__(32) __bf16 Bs[2][128 * 32];

  const int tid  = threadIdx.x;
  const int lane = tid & 31;
  const int wave = tid >> 5;
  const int m0 = blockIdx.x * 128;
  const int n0 = blockIdx.y * 128;

  const int l15  = lane & 15;
  const int half = lane >> 4;
  const int wm = wave >> 1;  // 0..3: 32-row group
  const int wn = wave & 1;   // 0..1: 64-col group

  // Staging: each thread owns 32B of A and of Wt per K-step.
  const int srow = tid >> 1;          // 0..127
  const int scol = (tid & 1) * 16;    // 0 or 16
  const __bf16* gA = &A[(size_t)(m0 + srow) * K + scol];
  const __bf16* gW = &Wt[(size_t)(n0 + srow) * K + scol];
  const uint32_t ldsA0 = (uint32_t)(uintptr_t)&As[0][srow * 32 + scol];
  const uint32_t ldsA1 = (uint32_t)(uintptr_t)&As[1][srow * 32 + scol];
  const uint32_t ldsB0 = (uint32_t)(uintptr_t)&Bs[0][srow * 32 + scol];
  const uint32_t ldsB1 = (uint32_t)(uintptr_t)&Bs[1][srow * 32 + scol];

  v8f acc[2][4];
#pragma unroll
  for (int a = 0; a < 2; ++a)
#pragma unroll
    for (int n = 0; n < 4; ++n) acc[a][n] = (v8f)0.f;

  // Prologue: stage tile 0 into buffer 0.
  async_copy32(ldsA0, gA);
  async_copy32(ldsB0, gW);
  wait_async0();
  __syncthreads();

  int p = 0;
  for (int k0 = 0; k0 < K; k0 += 32) {
    const bool more = (k0 + 32) < K;
    // Kick off the next tile's async stage immediately (other buffer).
    if (more) {
      async_copy32(p ? ldsA0 : ldsA1, gA + k0 + 32);
      async_copy32(p ? ldsB0 : ldsB1, gW + k0 + 32);
    }

    const __bf16* Asp = As[p];
    const __bf16* Bsp = Bs[p];

    // Two A fragments (rows wm*32 + a*16 + ...), contiguous k = 16*half + s, same as B.
    v16bf afr[2];
#pragma unroll
    for (int a = 0; a < 2; ++a) {
      const int arow = wm * 32 + a * 16 + l15;
      afr[a] = *(const v16bf*)&Asp[arow * 32 + half * 16];
    }

    // Four B fragments; each feeds two WMMAs immediately.
#pragma unroll
    for (int n = 0; n < 4; ++n) {
      const int brow = wn * 64 + n * 16 + l15;
      v16bf bfr = *(const v16bf*)&Bsp[brow * 32 + half * 16];
      acc[0][n] = wmma_bf16(afr[0], bfr, acc[0][n]);
      acc[1][n] = wmma_bf16(afr[1], bfr, acc[1][n]);
    }

    if (more) wait_async0();
    __syncthreads();
    p ^= 1;
  }

  // Epilogue. C layout: VGPR r, lanes 0-15 -> row r, lanes 16-31 -> row r+8.
#pragma unroll
  for (int a = 0; a < 2; ++a) {
#pragma unroll
    for (int n = 0; n < 4; ++n) {
#pragma unroll
      for (int r = 0; r < 8; ++r) {
        const int gm = m0 + wm * 32 + a * 16 + r + 8 * half;
        const int gn = n0 + wn * 64 + n * 16 + l15;
        const size_t o = (size_t)gm * N + gn;
        float v = acc[a][n][r];
        if (mode == 0) {
          outB[o] = (__bf16)v;
        } else if (mode == 1) {
          outF[o] = v + res[o];
        } else if (mode == 2) {
          float t = v + bias[gn];
          float g = 0.5f * t *
                    (1.f + tanhf(0.7978845608028654f * (t + 0.044715f * t * t * t)));
          outB[o] = (__bf16)g;
        } else {
          outF[o] = v + bias[gn] + res[o];
        }
      }
    }
  }
}

// ---------------------------------------------------------------------------
// Causal flash attention, one 16-row query tile per wave, 32-key chunks.
// Q,K: [BH, T, 64] bf16 ; Vt: [BH, 64, T] bf16 ; O: [B, T, H, 64] bf16.
// K-fragments batched ahead of the 4 score WMMAs; V-fragments hoisted above
// the softmax VALU work so global latency hides under it.
// ---------------------------------------------------------------------------
__global__ __launch_bounds__(256) void attn_kernel(const __bf16* __restrict__ Q,
                                                   const __bf16* __restrict__ K,
                                                   const __bf16* __restrict__ Vt,
                                                   __bf16* __restrict__ O,
                                                   int T) {
  __shared__ __align__(32) __bf16 Ps[8][16 * 32];  // per-wave P staging

  const int lane = threadIdx.x & 31;
  const int wave = threadIdx.x >> 5;
  const int l15  = lane & 15;
  const int half = lane >> 4;

  const int bh = blockIdx.y;
  const int b  = bh >> 4;
  const int h  = bh & 15;
  const int q0 = (blockIdx.x * 8 + wave) * 16;

  const __bf16* Qb = Q  + (size_t)bh * T * HD_;
  const __bf16* Kb = K  + (size_t)bh * T * HD_;
  const __bf16* Vb = Vt + (size_t)bh * HD_ * T;

  // Q fragments, kept resident (contraction hd: steps 0-31, 32-63).
  v16bf qf[2];
#pragma unroll
  for (int s = 0; s < 2; ++s) {
    qf[s] = *(const v16bf*)&Qb[(size_t)(q0 + l15) * HD_ + s * 32 + half * 16];
  }

  v8f o[4];
#pragma unroll
  for (int n = 0; n < 4; ++n) o[n] = (v8f)0.f;
  float mrow[8], lrow[8];
#pragma unroll
  for (int r = 0; r < 8; ++r) { mrow[r] = -1e30f; lrow[r] = 0.f; }

  const float scale = 0.125f;  // 1/sqrt(64)
  const int kend = q0 + 16;    // causal: keys <= q0+15

  for (int kb = 0; kb < kend; kb += 32) {
    if (kb + 32 < kend) {
      __builtin_prefetch(&Kb[(size_t)(kb + 32) * HD_], 0, 3);
      __builtin_prefetch(&Vb[kb + 32], 0, 3);
    }

    // ---- batch all 4 K-fragments, then 4 score WMMAs ----
    v16bf kf[2][2];
#pragma unroll
    for (int j = 0; j < 2; ++j)
#pragma unroll
      for (int s = 0; s < 2; ++s)
        kf[j][s] = *(const v16bf*)&Kb[(size_t)(kb + j * 16 + l15) * HD_ +
                                      s * 32 + half * 16];
    v8f st[2];
#pragma unroll
    for (int j = 0; j < 2; ++j) {
      v8f c = (v8f)0.f;
      c = wmma_bf16(qf[0], kf[j][0], c);
      c = wmma_bf16(qf[1], kf[j][1], c);
      st[j] = c;
    }

    // ---- hoist V-fragment loads above the softmax VALU work ----
    v16bf vf[4];
#pragma unroll
    for (int n = 0; n < 4; ++n)
      vf[n] = *(const v16bf*)&Vb[(size_t)(n * 16 + l15) * T + kb + half * 16];

    // ---- online softmax (rows split across lane halves per C layout) ----
    float p0[8], p1[8];
#pragma unroll
    for (int r = 0; r < 8; ++r) {
      const int gq = q0 + r + 8 * half;
      float x0 = st[0][r] * scale;
      float x1 = st[1][r] * scale;
      if (kb + l15 > gq)      x0 = -1e30f;
      if (kb + 16 + l15 > gq) x1 = -1e30f;
      float mx = fmaxf(x0, x1);
#pragma unroll
      for (int m = 1; m < 16; m <<= 1) mx = fmaxf(mx, __shfl_xor(mx, m, 32));
      const float mnew  = fmaxf(mrow[r], mx);
      const float alpha = __expf(mrow[r] - mnew);
      const float e0 = __expf(x0 - mnew);
      const float e1 = __expf(x1 - mnew);
      float ps = e0 + e1;
#pragma unroll
      for (int m = 1; m < 16; m <<= 1) ps += __shfl_xor(ps, m, 32);
      lrow[r] = lrow[r] * alpha + ps;
      mrow[r] = mnew;
      p0[r] = e0; p1[r] = e1;
#pragma unroll
      for (int n = 0; n < 4; ++n) o[n][r] *= alpha;
    }

    // ---- transpose P into A-matrix layout via per-wave LDS ----
    __bf16* P = Ps[wave];
#pragma unroll
    for (int r = 0; r < 8; ++r) {
      P[(r + 8 * half) * 32 + l15]      = (__bf16)p0[r];
      P[(r + 8 * half) * 32 + 16 + l15] = (__bf16)p1[r];
    }
    asm volatile("s_wait_dscnt 0" ::: "memory");
    v16bf pf = *(const v16bf*)&P[l15 * 32 + half * 16];

    // ---- O += P @ V ----
#pragma unroll
    for (int n = 0; n < 4; ++n) o[n] = wmma_bf16(pf, vf[n], o[n]);
  }

  // ---- normalize and store to [B, T, H, 64] bf16 ----
#pragma unroll
  for (int n = 0; n < 4; ++n) {
#pragma unroll
    for (int r = 0; r < 8; ++r) {
      const int gq = q0 + r + 8 * half;
      const float val = o[n][r] / lrow[r];
      O[((size_t)(b * T + gq) * H_ + h) * HD_ + n * 16 + l15] = (__bf16)val;
    }
  }
}

// ---------------------------------------------------------------------------
// Host-side launch. Inputs (setup_inputs order):
// 0 x, 1 positions, 2 mask, 3 ln1_scale, 4 ln1_bias, 5 w_qkv, 6 w_o,
// 7 ln2_scale, 8 ln2_bias, 9 w1, 10 b1, 11 w2, 12 b2.
// ---------------------------------------------------------------------------
extern "C" void kernel_launch(void* const* d_in, const int* in_sizes, int n_in,
                              void* d_out, int out_size, void* d_ws, size_t ws_size,
                              hipStream_t stream) {
  const float* x        = (const float*)d_in[0];
  const float* ln1s     = (const float*)d_in[3];
  const float* ln1b     = (const float*)d_in[4];
  const float* w_qkv    = (const float*)d_in[5];
  const float* w_o      = (const float*)d_in[6];
  const float* ln2s     = (const float*)d_in[7];
  const float* ln2b     = (const float*)d_in[8];
  const float* w1       = (const float*)d_in[9];
  const float* b1       = (const float*)d_in[10];
  const float* w2       = (const float*)d_in[11];
  const float* b2       = (const float*)d_in[12];
  float* out            = (float*)d_out;

  // Workspace layout (bytes), with lifetime-based aliasing (~104 MB total).
  char* ws = (char*)d_ws;
  __bf16* wqkvT = (__bf16*)(ws + 0);                    //  6 MB [3072,1024]
  __bf16* woT   = (__bf16*)(ws + 6291456);              //  2 MB [1024,1024]
  __bf16* w1T   = (__bf16*)(ws + 8388608);              //  8 MB [4096,1024]
  __bf16* w2T   = (__bf16*)(ws + 16777216);             //  8 MB [1024,4096]
  float*  x2    = (float*)(ws + 25165824);              // 16 MB [4096,1024]
  __bf16* hbf   = (__bf16*)(ws + 41943040);             //  8 MB (h, then attn out)
  __bf16* attnb = hbf;                                  //  alias (h dead post-QKV)
  __bf16* qkvb  = (__bf16*)(ws + 50331648);             // 24 MB (then mlp1, 32 MB)
  __bf16* mlp1b = qkvb;                                 //  alias (qkv dead post-RoPE/vtrans)
  __bf16* qb    = (__bf16*)(ws + 83886080);             //  8 MB (then h2)
  __bf16* h2b   = qb;                                   //  alias (q dead post-attn)
  __bf16* kb    = (__bf16*)(ws + 92274688);             //  8 MB
  __bf16* vtb   = (__bf16*)(ws + 100663296);            //  8 MB

  // 1) Weight conversions (f32 [K,N] -> bf16 [N,K]), LDS-tiled transpose.
  convT_kernel<<<dim3((3 * D_) / 32, D_ / 32), dim3(32, 8), 0, stream>>>(
      w_qkv, wqkvT, D_, 3 * D_);
  convT_kernel<<<dim3(D_ / 32, D_ / 32), dim3(32, 8), 0, stream>>>(
      w_o, woT, D_, D_);
  convT_kernel<<<dim3(HID_ / 32, D_ / 32), dim3(32, 8), 0, stream>>>(
      w1, w1T, D_, HID_);
  convT_kernel<<<dim3(D_ / 32, HID_ / 32), dim3(32, 8), 0, stream>>>(
      w2, w2T, HID_, D_);

  // 2) LN1: x -> h (bf16).
  ln_kernel<<<ROWS, 256, 0, stream>>>(x, ln1s, ln1b, hbf, D_);

  // 3) QKV GEMM: [4096,1024] x [1024,3072] -> bf16.
  gemm_bf16_kernel<<<dim3(ROWS / 128, (3 * D_) / 128), 256, 0, stream>>>(
      hbf, wqkvT, ROWS, 3 * D_, D_, 0, nullptr, nullptr, nullptr, qkvb);

  // 4) RoPE (Q,K) + tiled V transpose.
  rope_kernel<<<(B_ * T_ * H_ * 32) / 256, 256, 0, stream>>>(qkvb, qb, kb, T_);
  vtrans_kernel<<<dim3(T_ / 32, HD_ / 32, B_ * H_), dim3(32, 8), 0, stream>>>(
      qkvb, vtb, T_);

  // 5) Causal flash attention -> attn (bf16, [B,T,H,64] == [4096,1024]).
  attn_kernel<<<dim3(T_ / 128, B_ * H_), 256, 0, stream>>>(qb, kb, vtb, attnb, T_);

  // 6) O-proj + residual: x2 = attn @ w_o + x (f32).
  gemm_bf16_kernel<<<dim3(ROWS / 128, D_ / 128), 256, 0, stream>>>(
      attnb, woT, ROWS, D_, D_, 1, nullptr, x, x2, nullptr);

  // 7) LN2: x2 -> h2 (bf16).
  ln_kernel<<<ROWS, 256, 0, stream>>>(x2, ln2s, ln2b, h2b, D_);

  // 8) MLP1: gelu(h2 @ w1 + b1) -> bf16.
  gemm_bf16_kernel<<<dim3(ROWS / 128, HID_ / 128), 256, 0, stream>>>(
      h2b, w1T, ROWS, HID_, D_, 2, b1, nullptr, nullptr, mlp1b);

  // 9) MLP2 + bias + residual -> d_out (f32).
  gemm_bf16_kernel<<<dim3(ROWS / 128, D_ / 128), 256, 0, stream>>>(
      mlp1b, w2T, ROWS, D_, HID_, 3, b2, x2, out, nullptr);
}
